// StockFormer_57183194578947
// MI455X (gfx1250) — matrix operand order repaired and checked by reading, not compiled, not run on hardware
//
#include <hip/hip_runtime.h>
#include <hip/hip_bf16.h>
#include <math.h>

// ---------------------------------------------------------------------------
// StockFormer forward on MI455X (gfx1250, wave32).
// All GEMMs run on v_wmma_f32_16x16x32_f16 (f16 in, f32 accumulate).
// Workspace requirement: ~410 MB (activations staged in f16; two f32 buffers
// for the fused decoder outputs).
// ---------------------------------------------------------------------------

typedef __attribute__((ext_vector_type(16))) _Float16 v16h;
typedef __attribute__((ext_vector_type(8)))  _Float16 v8h;
typedef __attribute__((ext_vector_type(8)))  float    v8f;

#define DEVI static __device__ __forceinline__

// A-matrix fragment (16x32 f16): lane L holds row M=L%16;
// halves of the wave hold K = 16*h..16*h+15 contiguous (matches B).
DEVI v16h load_a_frag(const _Float16* row_kb, int half) {
  v8h c1 = *(const v8h*)(row_kb + 16 * half);
  v8h c2 = *(const v8h*)(row_kb + 16 * half + 8);
  v16h a;
#pragma unroll
  for (int i = 0; i < 8; ++i) { a[i] = c1[i]; a[8 + i] = c2[i]; }
  return a;
}

// B-matrix fragment (32x16 f16): lane L holds column N=L%16, K = 16*half..+15
// contiguous. For a Linear layer, B[k][n] = W[n][k] -> one 32B run per lane.
DEVI v16h load_b_frag(const _Float16* wrow_kb, int half) {
  return *(const v16h*)(wrow_kb + 16 * half);
}

DEVI v16h zero_frag() {
  v16h a;
#pragma unroll
  for (int i = 0; i < 16; ++i) a[i] = (_Float16)0.f;
  return a;
}

DEVI v8f wmma(v16h a, v16h b, v8f c) {
  return __builtin_amdgcn_wmma_f32_16x16x32_f16(false, a, false, b,
                                                (short)0, c, false, false);
}

// ---------------------------------------------------------------------------
// Weight staging
// ---------------------------------------------------------------------------
__global__ void f32_to_f16_k(const float* __restrict__ s, _Float16* __restrict__ d, int n) {
  int i = blockIdx.x * blockDim.x + threadIdx.x;
  if (i < n) d[i] = (_Float16)s[i];
}

// conv_w (O=128, IC=128, J=3) f32 -> f16 staged as [J][O][IC] (contiguous ic).
__global__ void conv_w_stage_k(const float* __restrict__ w, _Float16* __restrict__ d) {
  int i = blockIdx.x * blockDim.x + threadIdx.x;
  if (i < 128 * 128 * 3) {
    int j = i % 3, ic = (i / 3) % 128, o = i / 384;
    d[((size_t)j * 128 + o) * 128 + ic] = (_Float16)w[i];
  }
}

// ---------------------------------------------------------------------------
// Prep: Haar decouple + concat + positional encoding, emit f16 activations.
// xl16[q=b*128+n][s][0:64]=x+pe, [64:128]=x_l+pe ; xh16 same without pe.
// ---------------------------------------------------------------------------
__global__ void prep_k(const float* __restrict__ x,
                       _Float16* __restrict__ xl, _Float16* __restrict__ xh) {
  size_t i = (size_t)blockIdx.x * blockDim.x + threadIdx.x;  // over 4*256*128*64
  int f = (int)(i & 63);
  size_t r = i >> 6;
  int n = (int)(r & 127); r >>= 7;
  int s = (int)(r & 255);
  int b = (int)(r >> 8);
  float xv = x[i];
  float pv = x[i ^ 8192];                 // sequence pair partner (s ^ 1)
  float xlv = 0.5f * (xv + pv);
  float xhv = 0.5f * (xv - pv);           // uniform sign trick for stack([d,-d])
  const float c = 9.210340371976184f / 128.0f;   // ln(1e4)/d_model
  float fs = (float)s;
  float a1 = fs * __expf(-c * (float)(f & ~1));
  float a2 = fs * __expf(-c * (float)(64 + (f & ~1)));
  float pe1 = (f & 1) ? __cosf(a1) : __sinf(a1);
  float pe2 = (f & 1) ? __cosf(a2) : __sinf(a2);
  size_t q = (size_t)b * 128 + n;
  size_t base = (q * 256 + s) * 128;
  xl[base + f]      = (_Float16)(xv + pe1);
  xl[base + 64 + f] = (_Float16)(xlv + pe2);
  xh[base + f]      = (_Float16)xv;
  xh[base + 64 + f] = (_Float16)xhv;
}

// ---------------------------------------------------------------------------
// Generic Linear: out = A[M,K] @ W[N,K]^T + bias, WMMA tiled.
// Each wave owns a 32x32 output tile (2 A-frags x 2 B-frags -> 4 WMMAs per
// K-step, halving loads per WMMA); 4 waves/block -> 128 rows per block.
// ---------------------------------------------------------------------------
__global__ __launch_bounds__(128) void linear16_k(
    const _Float16* __restrict__ A, int lda,
    const _Float16* __restrict__ W, int ldw,
    const float* __restrict__ bias,
    _Float16* __restrict__ out16, float* __restrict__ out32, int ldo,
    int K, int relu) {
  int lane = threadIdx.x & 31, wave = threadIdx.x >> 5;
  int half = lane >> 4, l16 = lane & 15;
  int nb = blockIdx.x * 32;
  int mb = blockIdx.y * 128 + wave * 32;
  const _Float16* arow0 = A + (size_t)(mb + l16) * lda;
  const _Float16* arow1 = arow0 + (size_t)16 * lda;
  const _Float16* wrow0 = W + (size_t)(nb + l16) * ldw;
  const _Float16* wrow1 = wrow0 + (size_t)16 * ldw;
  v8f acc00 = {}, acc01 = {}, acc10 = {}, acc11 = {};
  for (int kb = 0; kb < K; kb += 32) {
    __builtin_prefetch(arow0 + kb + 128, 0, 1);   // global_prefetch_b8
    __builtin_prefetch(arow1 + kb + 128, 0, 1);
    v16h a0 = load_a_frag(arow0 + kb, half);
    v16h a1 = load_a_frag(arow1 + kb, half);
    v16h b0 = load_b_frag(wrow0 + kb, half);
    v16h b1 = load_b_frag(wrow1 + kb, half);
    acc00 = wmma(a0, b0, acc00);
    acc01 = wmma(a0, b1, acc01);
    acc10 = wmma(a1, b0, acc10);
    acc11 = wmma(a1, b1, acc11);
  }
  int n0 = nb + l16, n1 = n0 + 16;
  float bs0 = bias ? bias[n0] : 0.f;
  float bs1 = bias ? bias[n1] : 0.f;
#pragma unroll
  for (int r = 0; r < 8; ++r) {
    int m0 = mb + r + 8 * half, m1 = m0 + 16;
    float v00 = acc00[r] + bs0, v01 = acc01[r] + bs1;
    float v10 = acc10[r] + bs0, v11 = acc11[r] + bs1;
    if (relu) {
      v00 = fmaxf(v00, 0.f); v01 = fmaxf(v01, 0.f);
      v10 = fmaxf(v10, 0.f); v11 = fmaxf(v11, 0.f);
    }
    if (out16) {
      out16[(size_t)m0 * ldo + n0] = (_Float16)v00;
      out16[(size_t)m0 * ldo + n1] = (_Float16)v01;
      out16[(size_t)m1 * ldo + n0] = (_Float16)v10;
      out16[(size_t)m1 * ldo + n1] = (_Float16)v11;
    }
    if (out32) {
      out32[(size_t)m0 * ldo + n0] = v00;
      out32[(size_t)m0 * ldo + n1] = v01;
      out32[(size_t)m1 * ldo + n0] = v10;
      out32[(size_t)m1 * ldo + n1] = v11;
    }
  }
}

// ---------------------------------------------------------------------------
// Dilated conv1d (k=3, dil=2, pad=2) as 3 shifted WMMA GEMMs + relu,
// scattered into the reference's y_h = reshape/transpose layout.
// X: (512,256,128)[q][t][ic] f16, Wj: [3][o][ic] f16.
// ---------------------------------------------------------------------------
__global__ __launch_bounds__(128) void conv_wmma_k(
    const _Float16* __restrict__ X, const _Float16* __restrict__ Wj,
    const float* __restrict__ bias, _Float16* __restrict__ Yh) {
  int lane = threadIdx.x & 31, wave = threadIdx.x >> 5;
  int half = lane >> 4, l16 = lane & 15;
  int ob = blockIdx.x * 16;
  int tb = blockIdx.y * 64 + wave * 16;
  int q  = blockIdx.z;
  const _Float16* base = X + (size_t)q * 256 * 128;
  v8f acc = {};
  for (int j = 0; j < 3; ++j) {
    int tsrc = tb + l16 + 2 * j - 2;
    bool ok = (tsrc >= 0) && (tsrc < 256);
    const _Float16* arow = base + (size_t)(ok ? tsrc : 0) * 128;
    const _Float16* wrow = Wj + ((size_t)j * 128 + (ob + l16)) * 128;
    for (int kb = 0; kb < 128; kb += 32) {
      v16h a;
      if (ok) a = load_a_frag(arow + kb, half); else a = zero_frag();
      v16h b = load_b_frag(wrow + kb, half);
      acc = wmma(a, b, acc);
    }
  }
  int o = ob + l16;
  float bs = bias[o];
  int bidx = q >> 7, nc = q & 127;
#pragma unroll
  for (int r = 0; r < 8; ++r) {
    int sc = tb + r + 8 * half;
    float v = fmaxf(acc[r] + bs, 0.f);
    size_t qy = (size_t)bidx * 128 + (sc >> 1);
    size_t ty = (size_t)((sc & 1) * 128 + nc);
    Yh[(qy * 256 + ty) * 128 + o] = (_Float16)v;
  }
}

// ---------------------------------------------------------------------------
// MHA core: per (bn, head, 32-query chunk). hd=32 -> one WMMA per score tile.
// Scores in LDS f32, shuffle softmax, probs re-staged f16.
// V^T is cooperatively staged to LDS once per block so the P@V B-fragments
// become single contiguous 32B ds_load runs instead of 16 strided global
// scalar loads.
// ---------------------------------------------------------------------------
__global__ __launch_bounds__(64) void attn_wmma_k(
    const _Float16* __restrict__ QKV, int ld, int koff, int voff,
    _Float16* __restrict__ Out, float scale) {
  __shared__ float    sS[2 * 16 * 256];   // 32 KB scores (per-wave halves)
  __shared__ _Float16 sP[2 * 16 * 256];   // 16 KB probabilities
  __shared__ _Float16 sVT[32 * 256];      // 16 KB V^T[c][s], shared by block
  int lane = threadIdx.x & 31, wave = threadIdx.x >> 5;
  int half = lane >> 4, l16 = lane & 15;
  int h = blockIdx.y, q = blockIdx.z;
  int sb = blockIdx.x * 32 + wave * 16;
  float*    mS = sS + wave * 16 * 256;
  _Float16* mP = sP + wave * 16 * 256;
  const _Float16* baseQ = QKV + (size_t)q * 256 * ld + h * 32;
  const _Float16* baseK = baseQ + koff;
  const _Float16* baseV = baseQ + voff;

  // cooperative V^T staging: coalesced b128 row loads, transposed LDS scatter
  for (int s = threadIdx.x; s < 256; s += 64) {
    const _Float16* vrow = baseV + (size_t)s * ld;
#pragma unroll
    for (int c8 = 0; c8 < 4; ++c8) {
      v8h v = *(const v8h*)(vrow + 8 * c8);
#pragma unroll
      for (int e = 0; e < 8; ++e) sVT[(c8 * 8 + e) * 256 + s] = v[e];
    }
  }

  // Q fragment: 16 rows x 32 (whole head dim in one fragment)
  v16h aq = load_a_frag(baseQ + (size_t)(sb + l16) * ld, half);

  // scores = Q @ K^T * scale
  for (int nt = 0; nt < 16; ++nt) {
    v16h bk = load_b_frag(baseK + (size_t)(nt * 16 + l16) * ld, half);
    v8f sc = {};
    sc = wmma(aq, bk, sc);
#pragma unroll
    for (int r = 0; r < 8; ++r)
      mS[(r + 8 * half) * 256 + nt * 16 + l16] = sc[r] * scale;
  }
  __syncthreads();   // covers V^T staging + score visibility

  // row softmax (each wave owns its 16 rows; 32 lanes cover 256 cols)
  for (int row = 0; row < 16; ++row) {
    const float* rp = mS + row * 256;
    float mx = -3.0e38f;
#pragma unroll
    for (int i = 0; i < 8; ++i) mx = fmaxf(mx, rp[lane + 32 * i]);
    for (int off = 16; off; off >>= 1) mx = fmaxf(mx, __shfl_xor(mx, off, 32));
    float ev[8], sum = 0.f;
#pragma unroll
    for (int i = 0; i < 8; ++i) { ev[i] = __expf(rp[lane + 32 * i] - mx); sum += ev[i]; }
    for (int off = 16; off; off >>= 1) sum += __shfl_xor(sum, off, 32);
    float inv = 1.f / sum;
#pragma unroll
    for (int i = 0; i < 8; ++i) mP[row * 256 + lane + 32 * i] = (_Float16)(ev[i] * inv);
  }
  __syncthreads();

  // out = P(16x256) @ V(256x32): two 16-wide N tiles, K-loop of 32.
  // B-fragments now come from LDS V^T as contiguous 32B runs.
  v8f acc0 = {}, acc1 = {};
  for (int kb = 0; kb < 256; kb += 32) {
    v16h ap  = load_a_frag(mP + (size_t)l16 * 256 + kb, half);
    v16h bv0 = *(const v16h*)(sVT + (size_t)l16 * 256 + kb + 16 * half);
    v16h bv1 = *(const v16h*)(sVT + (size_t)(16 + l16) * 256 + kb + 16 * half);
    acc0 = wmma(ap, bv0, acc0);
    acc1 = wmma(ap, bv1, acc1);
  }
#pragma unroll
  for (int r = 0; r < 8; ++r) {
    size_t m = (size_t)q * 256 + sb + r + 8 * half;
    Out[m * 128 + h * 32 + l16]      = (_Float16)acc0[r];
    Out[m * 128 + h * 32 + 16 + l16] = (_Float16)acc1[r];
  }
}

// ---------------------------------------------------------------------------
// Head kernels (tiny linears + 2-way softmax). Flat row index m matches the
// reference's contiguous reshapes exactly.
// ---------------------------------------------------------------------------
__global__ void heads_local_k(const _Float16* __restrict__ y2,
                              const float* __restrict__ rlw, const float* __restrict__ rlb,
                              const float* __restrict__ clw, const float* __restrict__ clb,
                              float* __restrict__ out) {
  int m = blockIdx.x * blockDim.x + threadIdx.x;
  const _Float16* row = y2 + (size_t)m * 128;
  float r = rlb[0], c0 = clb[0], c1 = clb[1];
  for (int k = 0; k < 128; ++k) {
    float v = (float)row[k];
    r  = fmaf(v, rlw[k], r);
    c0 = fmaf(v, clw[k], c0);
    c1 = fmaf(v, clw[128 + k], c1);
  }
  out[393216 + m] = r;                                   // lreg
  float mx = fmaxf(c0, c1);
  float e0 = __expf(c0 - mx), e1 = __expf(c1 - mx);
  float inv = 1.f / (e0 + e1);
  out[524288 + 2 * m]     = e0 * inv;                    // lcla
  out[524288 + 2 * m + 1] = e1 * inv;
}

__global__ void heads_fused_k(const float* __restrict__ aself, const float* __restrict__ across,
                              const float* __restrict__ rfw, const float* __restrict__ rfb,
                              const float* __restrict__ cfw, const float* __restrict__ cfb,
                              float* __restrict__ out) {
  int m = blockIdx.x * blockDim.x + threadIdx.x;
  const float* ra = aself + (size_t)m * 128;
  const float* rb = across + (size_t)m * 128;
  float r = rfb[0], c0 = cfb[0], c1 = cfb[1];
  for (int k = 0; k < 128; ++k) {
    float v = ra[k] + rb[k];
    r  = fmaf(v, rfw[k], r);
    c0 = fmaf(v, cfw[k], c0);
    c1 = fmaf(v, cfw[128 + k], c1);
  }
  out[m] = r;                                            // reg
  float mx = fmaxf(c0, c1);
  float e0 = __expf(c0 - mx), e1 = __expf(c1 - mx);
  float inv = 1.f / (e0 + e1);
  out[131072 + 2 * m]     = e0 * inv;                    // cla
  out[131072 + 2 * m + 1] = e1 * inv;
}

// ---------------------------------------------------------------------------
extern "C" void kernel_launch(void* const* d_in, const int* in_sizes, int n_in,
                              void* d_out, int out_size, void* d_ws, size_t ws_size,
                              hipStream_t stream) {
  const float* x        = (const float*)d_in[0];
  /* d_in[1] = ts : dead code in the reference */
  const float* ta_w_in  = (const float*)d_in[2];
  const float* ta_b_in  = (const float*)d_in[3];
  const float* ta_w_out = (const float*)d_in[4];
  const float* ta_b_out = (const float*)d_in[5];
  const float* conv_w   = (const float*)d_in[6];
  const float* conv_b   = (const float*)d_in[7];
  const float* pl_w     = (const float*)d_in[8];
  const float* pl_b     = (const float*)d_in[9];
  const float* ph_w     = (const float*)d_in[10];
  const float* ph_b     = (const float*)d_in[11];
  const float* as_w_in  = (const float*)d_in[12];
  const float* as_b_in  = (const float*)d_in[13];
  const float* as_w_out = (const float*)d_in[14];
  const float* as_b_out = (const float*)d_in[15];
  const float* ac_w_in  = (const float*)d_in[16];
  const float* ac_b_in  = (const float*)d_in[17];
  const float* ac_w_out = (const float*)d_in[18];
  const float* ac_b_out = (const float*)d_in[19];
  const float* rl_w = (const float*)d_in[20];
  const float* rl_b = (const float*)d_in[21];
  const float* cl_w = (const float*)d_in[22];
  const float* cl_b = (const float*)d_in[23];
  const float* rf_w = (const float*)d_in[24];
  const float* rf_b = (const float*)d_in[25];
  const float* cf_w = (const float*)d_in[26];
  const float* cf_b = (const float*)d_in[27];
  float* out = (float*)d_out;

  const size_t MR = 131072;  // 512 sequences * 256 tokens
  char* ws = (char*)d_ws;
  size_t off = 0;
  auto alloc = [&](size_t bytes) -> char* {
    char* p = ws + off;
    off = (off + bytes + 255) & ~(size_t)255;
    return p;
  };

  // f16 weight staging
  _Float16* w_ta_in  = (_Float16*)alloc(49152 * 2);
  _Float16* w_ta_out = (_Float16*)alloc(16384 * 2);
  _Float16* w_pl     = (_Float16*)alloc(16384 * 2);
  _Float16* w_ph     = (_Float16*)alloc(16384 * 2);
  _Float16* w_as_in  = (_Float16*)alloc(49152 * 2);
  _Float16* w_as_out = (_Float16*)alloc(16384 * 2);
  _Float16* w_ac_in  = (_Float16*)alloc(49152 * 2);
  _Float16* w_ac_out = (_Float16*)alloc(16384 * 2);
  _Float16* w_conv   = (_Float16*)alloc(49152 * 2);

  // activation regions (reused across the sequential pipeline)
  _Float16* Rqkv = (_Float16*)alloc(MR * 384 * 2);
  _Float16* R1 = (_Float16*)alloc(MR * 128 * 2);  // xl16 -> q_l
  _Float16* R2 = (_Float16*)alloc(MR * 128 * 2);  // xh16 -> k_h
  _Float16* R3 = (_Float16*)alloc(MR * 128 * 2);  // attention outputs
  _Float16* R4 = (_Float16*)alloc(MR * 128 * 2);  // y_l -> y_l2
  _Float16* R5 = (_Float16*)alloc(MR * 128 * 2);  // y_h
  float* F1 = (float*)alloc(MR * 128 * 4);        // a_self
  float* F2 = (float*)alloc(MR * 128 * 4);        // a_cross

  auto cvt = [&](const float* s, _Float16* d, int n) {
    f32_to_f16_k<<<(n + 255) / 256, 256, 0, stream>>>(s, d, n);
  };
  cvt(ta_w_in, w_ta_in, 49152);
  cvt(ta_w_out, w_ta_out, 16384);
  cvt(pl_w, w_pl, 16384);
  cvt(ph_w, w_ph, 16384);
  cvt(as_w_in, w_as_in, 49152);
  cvt(as_w_out, w_as_out, 16384);
  cvt(ac_w_in, w_ac_in, 49152);
  cvt(ac_w_out, w_ac_out, 16384);
  conv_w_stage_k<<<(49152 + 255) / 256, 256, 0, stream>>>(conv_w, w_conv);

  // prep: haar + concat + pos-enc
  prep_k<<<8388608 / 256, 256, 0, stream>>>(x, R1, R2);

  auto linearW = [&](const _Float16* A, int lda, const _Float16* W, int ldw,
                     const float* bias, _Float16* o16, float* o32, int ldo,
                     int N, int K) {
    dim3 g(N / 32, MR / 128);
    linear16_k<<<g, 128, 0, stream>>>(A, lda, W, ldw, bias, o16, o32, ldo, K, 0);
  };

  const float scale = 0.17677669529663687f;  // 1/sqrt(32)

  // ---- temporal MHA (encoder low path) ----
  linearW(R1, 128, w_ta_in, 128, ta_b_in, Rqkv, nullptr, 384, 384, 128);
  attn_wmma_k<<<dim3(8, 4, 512), 64, 0, stream>>>(Rqkv, 384, 128 * 1, 128 * 2, R3, scale);
  linearW(R3, 128, w_ta_out, 128, ta_b_out, R4, nullptr, 128, 128, 128);  // y_l

  // ---- dilated conv (encoder high path) -> y_h (scattered layout) ----
  conv_wmma_k<<<dim3(8, 4, 512), 128, 0, stream>>>(R2, w_conv, conv_b, R5);

  // ---- decoder projections ----
  linearW(R4, 128, w_pl, 128, pl_b, R1, nullptr, 128, 128, 128);  // q_l
  linearW(R5, 128, w_ph, 128, ph_b, R2, nullptr, 128, 128, 128);  // k_h
  linearW(R1, 128, w_pl, 128, pl_b, R4, nullptr, 128, 128, 128);  // y_l2

  heads_local_k<<<512, 256, 0, stream>>>(R4, rl_w, rl_b, cl_w, cl_b, out);

  // ---- self attention on q_l ----
  linearW(R1, 128, w_as_in, 128, as_b_in, Rqkv, nullptr, 384, 384, 128);
  attn_wmma_k<<<dim3(8, 4, 512), 64, 0, stream>>>(Rqkv, 384, 128 * 1, 128 * 2, R3, scale);
  linearW(R3, 128, w_as_out, 128, as_b_out, nullptr, F1, 128, 128, 128);

  // ---- cross attention: q from q_l, k/v from k_h ----
  linearW(R1, 128, w_ac_in, 128, ac_b_in, Rqkv, nullptr, 384, 128, 128);           // Q proj
  linearW(R2, 128, w_ac_in + 128 * 128, 128, ac_b_in + 128, Rqkv + 128, nullptr,
          384, 256, 128);                                                          // K,V proj
  attn_wmma_k<<<dim3(8, 4, 512), 64, 0, stream>>>(Rqkv, 384, 128 * 1, 128 * 2, R3, scale);
  linearW(R3, 128, w_ac_out, 128, ac_b_out, nullptr, F2, 128, 128, 128);

  heads_fused_k<<<512, 256, 0, stream>>>(F1, F2, rf_w, rf_b, cf_w, cf_b, out);

  (void)in_sizes; (void)n_in; (void)out_size; (void)ws_size;
}
